// DiTAttention_85152021610602
// MI455X (gfx1250) — matrix operand order repaired and checked by reading, not compiled, not run on hardware
//
#include <hip/hip_runtime.h>

// ---------------------------------------------------------------------------
// Problem constants (match the JAX reference)
// ---------------------------------------------------------------------------
constexpr int B_   = 2;
constexpr int S_   = 2048;
constexpr int H_   = 2048;
constexpr int NH_  = 16;
constexpr int NKV_ = 4;
constexpr int HD_  = 128;
constexpr int NREP_ = NH_ / NKV_;       // 4
constexpr int M_   = B_ * S_;           // 4096 rows for all GEMMs
constexpr float SCALING_ = 0.08838834764831845f;  // 128^-0.5

// ---------------------------------------------------------------------------
// WMMA plumbing (CDNA5 / gfx1250, wave32)
// ---------------------------------------------------------------------------
typedef __attribute__((ext_vector_type(16))) __bf16 v16bf;
typedef __attribute__((ext_vector_type(8)))  __bf16 v8bf;
typedef __attribute__((ext_vector_type(8)))  float  v8f;
typedef __attribute__((ext_vector_type(4)))  unsigned int u32x4;
typedef __attribute__((ext_vector_type(8)))  int          i32x8;
typedef __attribute__((ext_vector_type(4)))  int          i32x4;

#define DEVINL static __device__ __forceinline__

// Tensor Data Mover availability (ROCm 7.2 clang-22: 5-arg builtin;
// amdgpu-toolchain clang-23: 6-arg builtin). Fallback = manual loads.
#if defined(__has_builtin)
#  if __has_builtin(__builtin_amdgcn_tensor_load_to_lds) && \
      __has_builtin(__builtin_amdgcn_s_wait_tensorcnt)
#    define HAS_TDM 1
#  endif
#endif
#ifndef HAS_TDM
#  define HAS_TDM 0
#endif

DEVINL v8f wmma_bf16(v16bf a, v16bf b, v8f c) {
  // D(f32 16x16) = A(bf16 16x32) * B(bf16 32x16) + C
  return __builtin_amdgcn_wmma_f32_16x16x32_bf16(
      /*neg_a=*/false, a, /*neg_b=*/false, b,
      /*c_mod=*/(short)0, c, /*reuse_a=*/false, /*reuse_b=*/false);
}

// Build a 16-element bf16 fragment from two 16-byte chunks.
DEVINL v16bf ld_frag(const __bf16* p0, const __bf16* p1) {
  union { v16bf v; v8bf h[2]; } u;
  u.h[0] = *(const v8bf*)p0;
  u.h[1] = *(const v8bf*)p1;
  return u.v;
}

DEVINL v8f vzero8() {
  v8f z;
#pragma unroll
  for (int i = 0; i < 8; ++i) z[i] = 0.0f;
  return z;
}

// Reductions across the 16-lane half-groups of a wave32 (C-matrix rows live
// in lanes 0-15 / 16-31; xor masks 1,2,4,8 stay within each half).
DEVINL float hmax16(float v) {
  v = fmaxf(v, __shfl_xor(v, 1, 32));
  v = fmaxf(v, __shfl_xor(v, 2, 32));
  v = fmaxf(v, __shfl_xor(v, 4, 32));
  v = fmaxf(v, __shfl_xor(v, 8, 32));
  return v;
}
DEVINL float hsum16(float v) {
  v += __shfl_xor(v, 1, 32);
  v += __shfl_xor(v, 2, 32);
  v += __shfl_xor(v, 4, 32);
  v += __shfl_xor(v, 8, 32);
  return v;
}

#if HAS_TDM
// ---------------------------------------------------------------------------
// Tensor Data Mover: DMA a 2D bf16 tile (rows x cols elements, row stride in
// elements) from global memory into LDS, with hardware row padding so the LDS
// stride comes out bank-conflict free.
//   pad_int_code: rows pad every 2^(code+1) DWORDs; pad_amt_code: pad (code+1)
//   DWORDs.  (ISA 8.4: D# group1 fields.)
// ---------------------------------------------------------------------------
DEVINL void tdm_load_2d(unsigned lds_off, const void* gptr,
                        unsigned rows, unsigned cols, unsigned stride_elems,
                        unsigned pad_int_code, unsigned pad_amt_code) {
  unsigned long long ga = (unsigned long long)(uintptr_t)gptr;
  u32x4 g0;
  g0[0] = 1u;                                   // count=1, user-mode descriptor
  g0[1] = lds_off;                              // lds_addr (bytes)
  g0[2] = (unsigned)(ga & 0xFFFFFFFFu);         // global_addr[31:0]
  g0[3] = (unsigned)((ga >> 32) & 0x01FFFFFFu)  // global_addr[56:32]
        | (2u << 30);                           // type = 2 ("image")
  i32x8 g1;
  g1[0] = (int)((1u << 16)                      // data_size = 2 bytes
              | (1u << 20)                      // pad_enable
              | ((pad_int_code & 7u) << 22)     // pad_interval
              | ((pad_amt_code & 127u) << 25)); // pad_amount
  g1[1] = (int)((cols & 0xFFFFu) << 16);        // tensor_dim0 (lo16)
  g1[2] = (int)((rows & 0xFFFFu) << 16);        // tensor_dim1 (lo16)
  g1[3] = (int)((cols & 0xFFFFu) << 16);        // tile_dim0
  g1[4] = (int)(rows & 0xFFFFu);                // tile_dim1 (tile_dim2 = 0)
  g1[5] = (int)stride_elems;                    // tensor_dim0_stride (lo32)
  g1[6] = 0;
  g1[7] = 0;
  i32x4 z4 = {0, 0, 0, 0};
#if __clang_major__ >= 23
  i32x8 z8 = {0, 0, 0, 0, 0, 0, 0, 0};
  __builtin_amdgcn_tensor_load_to_lds(g0, g1, z4, z4, z8, 0);
#else
  __builtin_amdgcn_tensor_load_to_lds(g0, g1, z4, z4, 0);
#endif
}
#endif  // HAS_TDM

// ---------------------------------------------------------------------------
// fp32 -> bf16 convert
// ---------------------------------------------------------------------------
__global__ void cvt_bf16_kernel(const float* __restrict__ src,
                                __bf16* __restrict__ dst, size_t n) {
  size_t i = (size_t)blockIdx.x * blockDim.x + threadIdx.x;
  if (i < n) dst[i] = (__bf16)src[i];
}

// ---------------------------------------------------------------------------
// Tiled bf16 WMMA GEMM:  C[M,N] = A[M,K] * B[K,N] + bias[N]   (A,B bf16 row-major)
// Block tile 128x128, 8 waves (4 along M x 2 along N), each wave 32x64.
// A tile staged by the Tensor Data Mover (pad -> LDT stride); B staged manually
// (needs a transpose, which TDM does not do).
// ---------------------------------------------------------------------------
constexpr int BM = 128, BN = 128, BK = 32;
constexpr int LDT = BK + 8;  // 40 elems = 80B rows -> conflict-free frag reads

__global__ __launch_bounds__(256)
void gemm_bf16_kernel(const __bf16* __restrict__ A, const __bf16* __restrict__ Bm,
                      const float* __restrict__ bias, float* __restrict__ C,
                      int M, int N, int K) {
  __shared__ __bf16 sA[BM * LDT];   // [row m][k]
  __shared__ __bf16 sB[BN * LDT];   // [col n][k]  (stored transposed)

  const int t    = threadIdx.x;
  const int lane = t & 31;
  const int wid  = t >> 5;
  const int l15  = lane & 15;
  const int hi   = lane >> 4;              // 0: lanes 0-15, 1: lanes 16-31
  const int bm   = blockIdx.y * BM;
  const int bn   = blockIdx.x * BN;
  const int wm   = (wid & 3) * 32;
  const int wn   = (wid >> 2) * 64;

  v8f acc[2][4];
#pragma unroll
  for (int mi = 0; mi < 2; ++mi)
#pragma unroll
    for (int ni = 0; ni < 4; ++ni) acc[mi][ni] = vzero8();

  const int bkr = t >> 3, bnc = (t & 7) * 16;     // B: 32 k x 128 n
#if HAS_TDM
  const unsigned sA_off = (unsigned)(uintptr_t)(void*)&sA[0];
#else
  const int ar = t >> 1, ac = (t & 1) * 16;       // A: 128 rows x 32 k
#endif

  for (int k0 = 0; k0 < K; k0 += BK) {
    // ---- stage A tile ----
#if HAS_TDM
    if (wid == 0) {
      // 128x32 bf16 tile; rows are 16 DWORDs -> pad_interval code 3; pad 4
      // DWORDs (8 bf16) -> LDS row stride = 40 elems = LDT.
      tdm_load_2d(sA_off, A + (size_t)bm * K + k0, BM, BK, (unsigned)K, 3u, 3u);
    }
#else
    {
      const __bf16* src = A + (size_t)(bm + ar) * K + (k0 + ac);
      *(v8bf*)&sA[ar * LDT + ac]     = *(const v8bf*)(src);
      *(v8bf*)&sA[ar * LDT + ac + 8] = *(const v8bf*)(src + 8);
    }
#endif
    // ---- stage B tile transposed ([n][k] in LDS) ----
    {
      const __bf16* src = Bm + (size_t)(k0 + bkr) * N + (bn + bnc);
      union { v8bf v; __bf16 e[8]; } u0, u1;
      u0.v = *(const v8bf*)(src);
      u1.v = *(const v8bf*)(src + 8);
#pragma unroll
      for (int i = 0; i < 8; ++i) {
        sB[(bnc + i)     * LDT + bkr] = u0.e[i];
        sB[(bnc + 8 + i) * LDT + bkr] = u1.e[i];
      }
    }
    // prefetch next B tile into cache (global_prefetch_b8)
    if (k0 + BK < K)
      __builtin_prefetch(Bm + (size_t)(k0 + BK + bkr) * N + (bn + bnc), 0, 3);
#if HAS_TDM
    if (wid == 0) __builtin_amdgcn_s_wait_tensorcnt(0);
#endif
    __syncthreads();

    // ---- fragments per ISA 7.12.2 layouts ----
    v16bf af[2], bfr[4];
#pragma unroll
    for (int mi = 0; mi < 2; ++mi) {
      const __bf16* p = &sA[(wm + mi * 16 + l15) * LDT + hi * 8];
      af[mi] = ld_frag(p, p + 16);            // K chunks {s0..s0+7, s0+16..s0+23}
    }
#pragma unroll
    for (int ni = 0; ni < 4; ++ni) {
      const __bf16* p = &sB[(wn + ni * 16 + l15) * LDT + hi * 8];
      bfr[ni] = ld_frag(p, p + 16);           // K chunks {s0..s0+7, s0+16..s0+23}
    }
#pragma unroll
    for (int mi = 0; mi < 2; ++mi)
#pragma unroll
      for (int ni = 0; ni < 4; ++ni)
        acc[mi][ni] = wmma_bf16(af[mi], bfr[ni], acc[mi][ni]);
    __syncthreads();
  }

  // ---- epilogue: C-layout row = j + 8*hi, col = l15 ----
#pragma unroll
  for (int mi = 0; mi < 2; ++mi) {
#pragma unroll
    for (int ni = 0; ni < 4; ++ni) {
      const int col = bn + wn + ni * 16 + l15;
      const float bv = bias[col];
#pragma unroll
      for (int j = 0; j < 8; ++j) {
        const int row = bm + wm + mi * 16 + j + hi * 8;
        C[(size_t)row * N + col] = acc[mi][ni][j] + bv;
      }
    }
  }
}

// ---------------------------------------------------------------------------
// RoPE + reorder: [B,S,nh,HD] fp32 -> [B,nh,S,HD] bf16 (softmax scale folded in)
// ---------------------------------------------------------------------------
__global__ void rope_reorder_kernel(const float* __restrict__ src,
                                    const float* __restrict__ cs,
                                    const float* __restrict__ sn,
                                    __bf16* __restrict__ dst,
                                    int nh, float scale) {
  size_t idx = (size_t)blockIdx.x * blockDim.x + threadIdx.x;
  size_t total = (size_t)B_ * S_ * nh * (HD_ / 2);
  if (idx >= total) return;
  int d = (int)(idx & 63);
  size_t r = idx >> 6;
  int hh = (int)(r % nh); r /= nh;
  int s  = (int)(r % S_);
  int b  = (int)(r / S_);
  const float* row  = src + ((size_t)(b * S_ + s) * nh + hh) * HD_;
  const float* crow = cs + (size_t)(b * S_ + s) * HD_;
  const float* srow = sn + (size_t)(b * S_ + s) * HD_;
  float x1 = row[d], x2 = row[d + 64];
  float o1 = (x1 * crow[d]      - x2 * srow[d])      * scale;
  float o2 = (x2 * crow[d + 64] + x1 * srow[d + 64]) * scale;
  __bf16* orow = dst + (((size_t)b * nh + hh) * S_ + s) * HD_;
  orow[d]      = (__bf16)o1;
  orow[d + 64] = (__bf16)o2;
}

// V reorder: [B,S,NKV,HD] fp32 -> [B,NKV,S,HD] bf16
__global__ void v_reorder_kernel(const float* __restrict__ src,
                                 __bf16* __restrict__ dst) {
  size_t idx = (size_t)blockIdx.x * blockDim.x + threadIdx.x;
  size_t total = (size_t)B_ * S_ * NKV_ * HD_;
  if (idx >= total) return;
  int d = (int)(idx % HD_); size_t r = idx / HD_;
  int hh = (int)(r % NKV_); r /= NKV_;
  int s  = (int)(r % S_);
  int b  = (int)(r / S_);
  dst[(((size_t)b * NKV_ + hh) * S_ + s) * HD_ + d] = (__bf16)src[idx];
}

// ---------------------------------------------------------------------------
// Flash attention (full softmax, GQA): 128 threads = 4 waves, 64 q-rows/block.
// Each wave owns 16 query rows; K tile fed by the Tensor Data Mover, V tile
// transposed manually into LDS.
// ---------------------------------------------------------------------------
constexpr int QT = 64, KT = 64;
constexpr int LDK = HD_ + 8;   // 136 elems = 272B rows -> conflict-free
constexpr int LDV = KT + 8;    // 72
constexpr int LDP = KT + 8;    // 72

__global__ __launch_bounds__(128)
void attn_kernel(const __bf16* __restrict__ Q,   // [B,NH,S,HD] (pre-scaled)
                 const __bf16* __restrict__ Kc,  // [B,NKV,S,HD]
                 const __bf16* __restrict__ Vc,  // [B,NKV,S,HD]
                 __bf16* __restrict__ O)         // [B,S,NH*HD]
{
  __shared__ __bf16 sK[KT * LDK];        // [key][d]
  __shared__ __bf16 sV[HD_ * LDV];       // [d][key]  (transposed)
  __shared__ __bf16 sP[4 * 16 * LDP];    // per-wave P scratch (C->A relayout)

  const int t    = threadIdx.x;
  const int lane = t & 31;
  const int wid  = t >> 5;
  const int l15  = lane & 15;
  const int hi   = lane >> 4;
  const int h  = blockIdx.y;
  const int b  = blockIdx.z;
  const int hk = h / NREP_;
  const int m0 = blockIdx.x * QT + wid * 16;

  // Q fragments: 16 rows x 128 d = 4 A-frags, loaded once.
  const __bf16* qbase = Q + ((size_t)(b * NH_ + h) * S_ + m0 + l15) * HD_;
  v16bf qf[4];
#pragma unroll
  for (int f = 0; f < 4; ++f) {
    const __bf16* p = qbase + f * 32 + hi * 8;
    qf[f] = ld_frag(p, p + 16);
  }

  v8f of[8];
#pragma unroll
  for (int dn = 0; dn < 8; ++dn) of[dn] = vzero8();
  float rm[8], rs[8];
#pragma unroll
  for (int j = 0; j < 8; ++j) { rm[j] = -3.0e38f; rs[j] = 0.0f; }

  const __bf16* kbase = Kc + (size_t)(b * NKV_ + hk) * S_ * HD_;
  const __bf16* vbase = Vc + (size_t)(b * NKV_ + hk) * S_ * HD_;
  __bf16* sPw = sP + wid * 16 * LDP;
#if HAS_TDM
  const unsigned sK_off = (unsigned)(uintptr_t)(void*)&sK[0];
#endif

  for (int kt = 0; kt < S_ / KT; ++kt) {
    const int kb = kt * KT;
    // ---- K-tile: TDM DMA, 64x128 bf16; rows are 64 DWORDs -> pad_interval
    //      code 5; pad 4 DWORDs (8 bf16) -> LDS row stride = 136 = LDK. ----
#if HAS_TDM
    if (wid == 0)
      tdm_load_2d(sK_off, kbase + (size_t)kb * HD_, KT, HD_, HD_, 5u, 3u);
#else
#pragma unroll
    for (int it = 0; it < 8; ++it) {
      int idx = t + it * 128;            // 1024 chunks of 8 bf16
      int row = idx >> 4, ch = idx & 15;
      *(v8bf*)&sK[row * LDK + ch * 8] =
          *(const v8bf*)(kbase + (size_t)(kb + row) * HD_ + ch * 8);
    }
#endif
    // ---- V-tile load, transposed into [d][key] (overlaps the TDM) ----
#pragma unroll
    for (int it = 0; it < 8; ++it) {
      int idx = t + it * 128;
      int key = idx >> 4, ch = idx & 15;
      union { v8bf v; __bf16 e[8]; } u;
      u.v = *(const v8bf*)(vbase + (size_t)(kb + key) * HD_ + ch * 8);
#pragma unroll
      for (int i = 0; i < 8; ++i)
        sV[(ch * 8 + i) * LDV + key] = u.e[i];
    }
#if HAS_TDM
    if (wid == 0) __builtin_amdgcn_s_wait_tensorcnt(0);
#endif
    __syncthreads();

    // ---- S = Q * K^T  (4 key sub-tiles of 16, 4 WMMAs each over d) ----
    v8f sf[4];
#pragma unroll
    for (int kn = 0; kn < 4; ++kn) {
      v8f s = vzero8();
#pragma unroll
      for (int f = 0; f < 4; ++f) {
        const __bf16* p = &sK[(kn * 16 + l15) * LDK + f * 32 + hi * 8];
        s = wmma_bf16(qf[f], ld_frag(p, p + 16), s);
      }
      sf[kn] = s;
    }

    // ---- online softmax update ----
    float corr[8];
#pragma unroll
    for (int j = 0; j < 8; ++j) {
      float tm = fmaxf(fmaxf(sf[0][j], sf[1][j]), fmaxf(sf[2][j], sf[3][j]));
      tm = hmax16(tm);
      float mnew = fmaxf(rm[j], tm);
      corr[j] = __expf(rm[j] - mnew);
      rm[j] = mnew;
    }
    const int prow = hi * 8;
#pragma unroll
    for (int j = 0; j < 8; ++j) {
      float psum = 0.0f;
#pragma unroll
      for (int kn = 0; kn < 4; ++kn) {
        float p = __expf(sf[kn][j] - rm[j]);
        psum += p;
        sPw[(prow + j) * LDP + kn * 16 + l15] = (__bf16)p;   // C-layout -> LDS
      }
      psum = hsum16(psum);
      rs[j] = rs[j] * corr[j] + psum;
#pragma unroll
      for (int dn = 0; dn < 8; ++dn) of[dn][j] *= corr[j];
    }

    // ---- O += P * V  (re-read P in A-frag layout; V^T gives contiguous K) ----
#pragma unroll
    for (int step = 0; step < 2; ++step) {
      const __bf16* pp = &sPw[l15 * LDP + step * 32 + hi * 8];
      v16bf pf = ld_frag(pp, pp + 16);
#pragma unroll
      for (int dn = 0; dn < 8; ++dn) {
        const __bf16* vp = &sV[(dn * 16 + l15) * LDV + step * 32 + hi * 8];
        of[dn] = wmma_bf16(pf, ld_frag(vp, vp + 16), of[dn]);
      }
    }
    __syncthreads();   // protect sK/sV before next tile overwrite
  }

  // ---- normalize and store to [B,S,NH*HD] bf16 for the O-projection GEMM ----
#pragma unroll
  for (int j = 0; j < 8; ++j) {
    const int srow = m0 + j + hi * 8;
    const float inv = 1.0f / rs[j];
    __bf16* orow = O + (size_t)(b * S_ + srow) * (NH_ * HD_) + h * HD_;
#pragma unroll
    for (int dn = 0; dn < 8; ++dn)
      orow[dn * 16 + l15] = (__bf16)(of[dn][j] * inv);
  }
}

// ---------------------------------------------------------------------------
// Launcher
// ---------------------------------------------------------------------------
extern "C" void kernel_launch(void* const* d_in, const int* in_sizes, int n_in,
                              void* d_out, int out_size, void* d_ws, size_t ws_size,
                              hipStream_t stream) {
  (void)in_sizes; (void)n_in; (void)out_size; (void)ws_size;
  const float* hidden = (const float*)d_in[0];
  const float* cosp   = (const float*)d_in[1];
  const float* sinp   = (const float*)d_in[2];
  const float* Wq = (const float*)d_in[3];
  const float* bq = (const float*)d_in[4];
  const float* Wk = (const float*)d_in[5];
  const float* bk = (const float*)d_in[6];
  const float* Wv = (const float*)d_in[7];
  const float* bv = (const float*)d_in[8];
  const float* Wo = (const float*)d_in[9];
  const float* bo = (const float*)d_in[10];
  float* out = (float*)d_out;

  char* base = (char*)d_ws;
  size_t off = 0;
  auto take = [&](size_t bytes) -> char* {
    char* p = base + off;
    off += (bytes + 255) & ~(size_t)255;
    return p;
  };
  __bf16* Xbf = (__bf16*)take((size_t)M_ * H_ * 2);
  __bf16* Wqb = (__bf16*)take((size_t)H_ * NH_ * HD_ * 2);
  __bf16* Wkb = (__bf16*)take((size_t)H_ * NKV_ * HD_ * 2);
  __bf16* Wvb = (__bf16*)take((size_t)H_ * NKV_ * HD_ * 2);
  __bf16* Wob = (__bf16*)take((size_t)NH_ * HD_ * H_ * 2);
  float*  qt  = (float*)take((size_t)M_ * NH_ * HD_ * 4);
  float*  ktf = (float*)take((size_t)M_ * NKV_ * HD_ * 4);
  float*  vtf = (float*)take((size_t)M_ * NKV_ * HD_ * 4);
  __bf16* Qb  = (__bf16*)take((size_t)M_ * NH_ * HD_ * 2);
  __bf16* Kb  = (__bf16*)take((size_t)M_ * NKV_ * HD_ * 2);
  __bf16* Vb  = (__bf16*)take((size_t)M_ * NKV_ * HD_ * 2);
  __bf16* Ab  = Xbf;   // X is dead after the QKV GEMMs; reuse for attn output

  const int thr = 256;
  auto nb = [](size_t n, int t) { return (unsigned)((n + t - 1) / t); };

  // 1) convert activations & weights to bf16
  size_t nX = (size_t)M_ * H_;
  cvt_bf16_kernel<<<nb(nX, thr), thr, 0, stream>>>(hidden, Xbf, nX);
  size_t nWq = (size_t)H_ * NH_ * HD_;
  cvt_bf16_kernel<<<nb(nWq, thr), thr, 0, stream>>>(Wq, Wqb, nWq);
  size_t nWk = (size_t)H_ * NKV_ * HD_;
  cvt_bf16_kernel<<<nb(nWk, thr), thr, 0, stream>>>(Wk, Wkb, nWk);
  cvt_bf16_kernel<<<nb(nWk, thr), thr, 0, stream>>>(Wv, Wvb, nWk);
  size_t nWo = (size_t)NH_ * HD_ * H_;
  cvt_bf16_kernel<<<nb(nWo, thr), thr, 0, stream>>>(Wo, Wob, nWo);

  // 2) QKV projections (bias fused)
  dim3 gq((NH_ * HD_) / BN, M_ / BM);
  gemm_bf16_kernel<<<gq, 256, 0, stream>>>(Xbf, Wqb, bq, qt, M_, NH_ * HD_, H_);
  dim3 gkv((NKV_ * HD_) / BN, M_ / BM);
  gemm_bf16_kernel<<<gkv, 256, 0, stream>>>(Xbf, Wkb, bk, ktf, M_, NKV_ * HD_, H_);
  gemm_bf16_kernel<<<gkv, 256, 0, stream>>>(Xbf, Wvb, bv, vtf, M_, NKV_ * HD_, H_);

  // 3) RoPE (+ fold softmax scale into Q) and head-major relayout to bf16
  size_t nQr = (size_t)B_ * S_ * NH_ * (HD_ / 2);
  rope_reorder_kernel<<<nb(nQr, thr), thr, 0, stream>>>(qt, cosp, sinp, Qb, NH_, SCALING_);
  size_t nKr = (size_t)B_ * S_ * NKV_ * (HD_ / 2);
  rope_reorder_kernel<<<nb(nKr, thr), thr, 0, stream>>>(ktf, cosp, sinp, Kb, NKV_, 1.0f);
  size_t nV = (size_t)B_ * S_ * NKV_ * HD_;
  v_reorder_kernel<<<nb(nV, thr), thr, 0, stream>>>(vtf, Vb);

  // 4) flash attention
  dim3 ga(S_ / QT, NH_, B_);
  attn_kernel<<<ga, 128, 0, stream>>>(Qb, Kb, Vb, Ab);

  // 5) output projection straight into d_out (fp32 + bias)
  dim3 go(H_ / BN, M_ / BM);
  gemm_bf16_kernel<<<go, 256, 0, stream>>>(Ab, Wob, bo, out, M_, H_, NH_ * HD_);
}
